// DGCNN_67130338836686
// MI455X (gfx1250) — matrix operand order repaired and checked by reading, not compiled, not run on hardware
//
#include <hip/hip_runtime.h>
#include <hip/hip_bf16.h>
#include <math.h>

typedef __attribute__((ext_vector_type(16))) _Float16 v16h;
typedef __attribute__((ext_vector_type(8)))  float    v8f;
typedef __attribute__((ext_vector_type(4)))  float    v4f;

#define BGR   128
#define NPER  512
#define EPER  512
#define DEG   32
#define FDIM  128
#define KTOP  30
#define TLD   97
#define HCS   104                 /* physical row stride of hcat (16B-aligned rows) */
#define NTOT  (BGR*NPER)          /* 65536 */
#define ETOT  (BGR*EPER)          /* 65536 */
#define NNZT  ((long long)NTOT*DEG)
#define C1K   16
#define C2K   32
#define KW2K  5
#define L1LEN 30
#define L2LEN 15
#define L3LEN 11
#define DENSE (C2K*L3LEN)         /* 352 */
#define ODIM  2

// ---------------- elementwise / prep kernels ----------------

__global__ __launch_bounds__(256) void k_fill(float* p, float v, long long n) {
  long long i = (long long)blockIdx.x * blockDim.x + threadIdx.x;
  if (i < n) p[i] = v;
}

__global__ __launch_bounds__(256) void k_count(const int* __restrict__ ie, float* hs, long long n) {
  long long i = (long long)blockIdx.x * blockDim.x + threadIdx.x;
  if (i < n) atomicAdd(&hs[ie[i]], 1.0f);
}

__global__ __launch_bounds__(256) void k_inv(float* p, long long n) {
  long long i = (long long)blockIdx.x * blockDim.x + threadIdx.x;
  if (i < n) p[i] = 1.0f / p[i];
}

// Pack weights [K,dout] f32 -> B fragments in exact WMMA lane order, zero-padded.
// Layout: Wfrag[((ct*ksteps + ks)*32 + lane)*16 + e],  k = ks*32 + ((e&8)<<1) + ((lane>>4)<<3) + (e&7),
//         c = ct*16 + (lane&15)
__global__ __launch_bounds__(256) void k_pack(const float* __restrict__ w, _Float16* __restrict__ wp,
                                              int K, int dout, int ksteps, long long total) {
  long long i = (long long)blockIdx.x * blockDim.x + threadIdx.x;
  if (i >= total) return;
  int e    = (int)(i & 15);
  int lane = (int)((i >> 4) & 31);
  long long rest = i >> 9;
  int ks = (int)(rest % ksteps);
  int ct = (int)(rest / ksteps);
  int k = ks * 32 + ((e & 8) << 1) + ((lane >> 4) << 3) + (e & 7);
  int c = ct * 16 + (lane & 15);
  float v = (k < K && c < dout) ? w[(long long)k * dout + c] : 0.0f;
  wp[i] = (_Float16)v;
}

// ---------------- WMMA GEMM: Y[R,Nact] = f16(X[R,Kpad]) @ W  (K pre-padded) ----------
// One wave per 16x16 output tile; unconditional aligned b128 loads, no divergence.

__global__ __launch_bounds__(256) void k_gemm_wmma(
    const float* __restrict__ X, int ldX, int Kpad,
    const _Float16* __restrict__ Wfrag,
    float* __restrict__ Y, int ldY, int Nact, int ntiles)
{
  int tid    = blockIdx.x * blockDim.x + threadIdx.x;
  int wave   = tid >> 5;
  int lane   = tid & 31;
  int tile_m = wave / ntiles;
  int tile_n = wave % ntiles;
  int m      = lane & 15;
  int hi     = lane >> 4;        // 0: lanes 0-15, 1: lanes 16-31
  int hi8    = hi << 3;
  int ksteps = Kpad >> 5;

  const float* xrow  = X + (long long)(tile_m * 16 + m) * ldX + hi8;
  const v16h*  wbase = (const v16h*)Wfrag + (long long)tile_n * ksteps * 32 + lane;

  v8f acc = {};
  for (int ks = 0; ks < ksteps; ++ks) {
    const v4f* xp = (const v4f*)(xrow + ks * 32);
    v4f x0 = xp[0];   // k = kk+hi8+0..3   -> a[0..3]
    v4f x1 = xp[1];   // k = kk+hi8+4..7   -> a[4..7]
    v4f x2 = xp[4];   // k = kk+16+hi8+0..3-> a[8..11]
    v4f x3 = xp[5];   // k = kk+16+hi8+4..7-> a[12..15]
    v16h a;
#pragma unroll
    for (int j = 0; j < 4; ++j) {
      a[j]      = (_Float16)x0[j];
      a[4 + j]  = (_Float16)x1[j];
      a[8 + j]  = (_Float16)x2[j];
      a[12 + j] = (_Float16)x3[j];
    }
    v16h bf = wbase[(long long)ks * 32];
    acc = __builtin_amdgcn_wmma_f32_16x16x32_f16(false, a, false, bf, (short)0, acc, false, false);
  }

  int col = tile_n * 16 + m;
  if (col < Nact) {
#pragma unroll
    for (int r = 0; r < 8; ++r) {
      long long rr = (long long)tile_m * 16 + r + hi8;   // C/D layout: vgpr r -> M=r / r+8
      Y[rr * ldY + col] = acc[r];
    }
  }
}

// ---------------- message passing ----------------

__global__ __launch_bounds__(256) void k_scatter(const float* __restrict__ t,
                                                 const int* __restrict__ in_,
                                                 const int* __restrict__ ie,
                                                 float* agg, int d, long long total) {
  long long i = (long long)blockIdx.x * blockDim.x + threadIdx.x;
  if (i >= total) return;
  long long nz = i / d;
  int c = (int)(i - nz * d);
  int n = in_[nz];
  int e = ie[nz];
  atomicAdd(&agg[(long long)e * d + c], t[(long long)n * d + c]);
}

// agg[E,dE] -> hE[E,32] zero-padded (so downstream GEMM needs no K bounds checks)
__global__ __launch_bounds__(256) void k_edge_act(const float* __restrict__ agg,
                                                  const float* __restrict__ bias,
                                                  const float* __restrict__ inv,
                                                  float* __restrict__ out, int dE, long long total) {
  long long i = (long long)blockIdx.x * blockDim.x + threadIdx.x;
  if (i >= total) return;
  long long e = i >> 5;
  int c = (int)(i & 31);
  float v = 0.0f;
  if (c < dE) v = tanhf((agg[e * dE + c] + bias[c]) * inv[e]);
  out[i] = v;
}

__global__ __launch_bounds__(256) void k_gather(const float* __restrict__ u,
                                                const int* __restrict__ ie,
                                                float* __restrict__ agg, int d, long long total) {
  long long i = (long long)blockIdx.x * blockDim.x + threadIdx.x;
  if (i >= total) return;
  long long n = i / d;
  int c = (int)(i - n * d);
  long long base = n * DEG;
  float s = 0.0f;
#pragma unroll 4
  for (int j = 0; j < DEG; ++j)
    s += u[(long long)ie[base + j] * d + c];
  agg[i] = s;
}

__global__ __launch_bounds__(256) void k_node_act(const float* __restrict__ agg,
                                                  const float* __restrict__ bias,
                                                  float* __restrict__ hcat, int off, int d, long long total) {
  long long i = (long long)blockIdx.x * blockDim.x + threadIdx.x;
  if (i >= total) return;
  long long n = i / d;
  int c = (int)(i - n * d);
  hcat[n * HCS + off + c] = tanhf((agg[i] + bias[c]) * (1.0f / 33.0f));  // node hdeg == DEG+1 == 33
}

// ---------------- sort-pool + conv tower ----------------

__global__ __launch_bounds__(256) void k_topk(const float* __restrict__ hcat, int* __restrict__ idx) {
  int b = blockIdx.x * blockDim.x + threadIdx.x;
  if (b >= BGR) return;
  unsigned long long used[8] = {0, 0, 0, 0, 0, 0, 0, 0};
  for (int kk = 0; kk < KTOP; ++kk) {
    float best = -INFINITY;
    int bi = 0;
    for (int j = 0; j < NPER; ++j) {
      if ((used[j >> 6] >> (j & 63)) & 1ull) continue;
      float v = hcat[((long long)b * NPER + j) * HCS + (TLD - 1)];
      if (v > best) { best = v; bi = j; }        // strict > : ties keep lower index (top_k stable)
    }
    used[bi >> 6] |= 1ull << (bi & 63);
    idx[b * KTOP + kk] = bi;
  }
}

__global__ __launch_bounds__(256) void k_pool(const float* __restrict__ hcat, const int* __restrict__ idx,
                                              float* __restrict__ pooled, long long total) {
  long long i = (long long)blockIdx.x * blockDim.x + threadIdx.x;
  if (i >= total) return;
  int b  = (int)(i / (KTOP * TLD));
  int r  = (int)(i - (long long)b * (KTOP * TLD));
  int kk = r / TLD;
  int f  = r - kk * TLD;
  int j  = idx[b * KTOP + kk];
  pooled[i] = hcat[((long long)b * NPER + j) * HCS + f];
}

__global__ __launch_bounds__(256) void k_conv1(const float* __restrict__ pooled,
                                               const float* __restrict__ w, const float* __restrict__ bb,
                                               float* __restrict__ y, long long total) {
  long long i = (long long)blockIdx.x * blockDim.x + threadIdx.x;
  if (i >= total) return;
  int b = (int)(i / (C1K * L1LEN));
  int r = (int)(i - (long long)b * (C1K * L1LEN));
  int c = r / L1LEN, t = r - c * L1LEN;
  const float* xp = pooled + (long long)b * KTOP * TLD + (long long)t * TLD;  // stride TLD, window TLD
  const float* wp = w + (long long)c * TLD;
  float s = bb[c];
  for (int j = 0; j < TLD; ++j) s += xp[j] * wp[j];
  y[i] = fmaxf(s, 0.0f);
}

__global__ __launch_bounds__(256) void k_mpool(const float* __restrict__ y1, float* __restrict__ y2, long long total) {
  long long i = (long long)blockIdx.x * blockDim.x + threadIdx.x;
  if (i >= total) return;
  int b = (int)(i / (C1K * L2LEN));
  int r = (int)(i - (long long)b * (C1K * L2LEN));
  int c = r / L2LEN, p = r - c * L2LEN;
  const float* yp = y1 + (long long)b * C1K * L1LEN + (long long)c * L1LEN + 2 * p;
  y2[i] = fmaxf(yp[0], yp[1]);
}

__global__ __launch_bounds__(256) void k_conv2(const float* __restrict__ y2,
                                               const float* __restrict__ w, const float* __restrict__ bb,
                                               float* __restrict__ y3, long long total) {
  long long i = (long long)blockIdx.x * blockDim.x + threadIdx.x;
  if (i >= total) return;
  int b = (int)(i / (C2K * L3LEN));
  int r = (int)(i - (long long)b * (C2K * L3LEN));
  int c = r / L3LEN, t = r - c * L3LEN;
  float s = bb[c];
  for (int ci = 0; ci < C1K; ++ci) {
    const float* yp = y2 + (long long)b * C1K * L2LEN + (long long)ci * L2LEN + t;
    const float* wp = w + (long long)c * C1K * KW2K + (long long)ci * KW2K;
#pragma unroll
    for (int kk = 0; kk < KW2K; ++kk) s += yp[kk] * wp[kk];
  }
  y3[i] = fmaxf(s, 0.0f);
}

__global__ __launch_bounds__(256) void k_dense(const float* __restrict__ y3,
                                               const float* __restrict__ w, const float* __restrict__ bb,
                                               float* __restrict__ out, int total) {
  int i = blockIdx.x * blockDim.x + threadIdx.x;
  if (i >= total) return;
  int b = i / ODIM, o = i - b * ODIM;
  const float* yp = y3 + (long long)b * DENSE;
  float s = bb[o];
  for (int j = 0; j < DENSE; ++j) s += yp[j] * w[(long long)j * ODIM + o];
  out[i] = fmaxf(s, 0.0f);   // relu(relu(x)) == relu(x)
}

// ---------------- host side ----------------

static inline int cdiv256(long long n) { return (int)((n + 255) / 256); }
#define LAUNCH(kern, n, ...) kern<<<cdiv256((long long)(n)), 256, 0, stream>>>(__VA_ARGS__)

static void run_gemm(const float* X, int ldX, int R, int Kpad, const _Float16* Wfrag, int dout,
                     float* Y, hipStream_t stream) {
  int ntiles = (dout + 15) / 16;
  long long threads = (long long)(R / 16) * ntiles * 32;   // R=65536 -> divisible by 256
  k_gemm_wmma<<<(int)(threads / 256), 256, 0, stream>>>(X, ldX, Kpad, Wfrag, Y, dout, dout, ntiles);
}

extern "C" void kernel_launch(void* const* d_in, const int* in_sizes, int n_in,
                              void* d_out, int out_size, void* d_ws, size_t ws_size,
                              hipStream_t stream) {
  (void)in_sizes; (void)n_in; (void)out_size; (void)ws_size;

  const float* node_feat = (const float*)d_in[0];
  const int*   inc_node  = (const int*)d_in[1];
  const int*   inc_edge  = (const int*)d_in[2];
  const float* W[8]; const float* Bv[8];
  for (int i = 0; i < 8; ++i) { W[i] = (const float*)d_in[3 + 2 * i]; Bv[i] = (const float*)d_in[4 + 2 * i]; }
  const float* c1w = (const float*)d_in[19];
  const float* c1b = (const float*)d_in[20];
  const float* c2w = (const float*)d_in[21];
  const float* c2b = (const float*)d_in[22];
  const float* ow  = (const float*)d_in[23];
  const float* ob  = (const float*)d_in[24];

  // bump allocator over workspace
  char* ws = (char*)d_ws;
  size_t off = 0;
  auto alloc = [&](size_t bytes) -> void* {
    void* p = ws + off;
    off += (bytes + 255) & ~(size_t)255;
    return p;
  };
  float* t      = (float*)alloc((size_t)NTOT * 32 * 4);   // projection buffer (N==E rows)
  float* aggE   = (float*)alloc((size_t)ETOT * 32 * 4);
  float* hE     = (float*)alloc((size_t)ETOT * 32 * 4);   // always written 32-wide, zero-padded
  float* aggN   = (float*)alloc((size_t)NTOT * 32 * 4);
  float* hcat   = (float*)alloc((size_t)NTOT * HCS * 4);  // stride-104 rows, 97 live channels
  float* hsize  = (float*)alloc((size_t)ETOT * 4);        // becomes 1/hsize
  _Float16* wp[8];
  for (int i = 0; i < 8; ++i) wp[i] = (_Float16*)alloc((size_t)128 * 32 * 2);
  int*   idxb   = (int*)alloc((size_t)BGR * KTOP * 4);
  float* pooled = (float*)alloc((size_t)BGR * KTOP * TLD * 4);
  float* y1     = (float*)alloc((size_t)BGR * C1K * L1LEN * 4);
  float* y2     = (float*)alloc((size_t)BGR * C1K * L2LEN * 4);
  float* y3     = (float*)alloc((size_t)BGR * C2K * L3LEN * 4);

  // hyperedge sizes: hsize[e] = 1 + count, then invert (exact fp32 counts -> deterministic)
  LAUNCH(k_fill, ETOT, hsize, 1.0f, (long long)ETOT);
  LAUNCH(k_count, NNZT, inc_edge, hsize, NNZT);
  LAUNCH(k_inv, ETOT, hsize, (long long)ETOT);

  // pack weights into WMMA B-fragment order (zero-padded K and N)
  const int wK[8] = {128, 32, 32, 32, 32, 32, 32, 1};
  const int wD[8] = {32, 32, 32, 32, 32, 32, 1, 1};
  for (int i = 0; i < 8; ++i) {
    int ntiles = (wD[i] + 15) / 16;
    int ksteps = ((wK[i] + 31) & ~31) >> 5;
    long long tot = (long long)ntiles * ksteps * 32 * 16;
    LAUNCH(k_pack, tot, W[i], wp[i], wK[i], wD[i], ksteps, tot);
  }

  const int offc[4] = {0, 32, 64, 96};
  for (int L = 0; L < 4; ++L) {
    int iE = 2 * L, iN = 2 * L + 1;
    int Kpad         = (L == 0) ? FDIM : 32;
    const float* X   = (L == 0) ? node_feat : (hcat + offc[L - 1]);
    int ldX          = (L == 0) ? FDIM : HCS;
    int dE = wD[iE], dN = wD[iN];

    // project-then-scatter: t = h_node @ wE, then segment-sum over inc_edge
    run_gemm(X, ldX, NTOT, Kpad, wp[iE], dE, t, stream);
    LAUNCH(k_fill, (long long)ETOT * dE, aggE, 0.0f, (long long)ETOT * dE);
    LAUNCH(k_scatter, NNZT * dE, t, inc_node, inc_edge, aggE, dE, NNZT * dE);
    LAUNCH(k_edge_act, (long long)ETOT * 32, aggE, Bv[iE], hsize, hE, dE, (long long)ETOT * 32);

    // project-then-gather: u = h_edge @ wN (hE is zero-padded to 32), contiguous per-node gather
    run_gemm(hE, 32, ETOT, 32, wp[iN], dN, t, stream);
    LAUNCH(k_gather, (long long)NTOT * dN, t, inc_edge, aggN, dN, (long long)NTOT * dN);
    LAUNCH(k_node_act, (long long)NTOT * dN, aggN, Bv[iN], hcat, offc[L], dN, (long long)NTOT * dN);
  }

  // sort-pool + conv tower
  LAUNCH(k_topk, BGR, hcat, idxb);
  long long ptot = (long long)BGR * KTOP * TLD;
  LAUNCH(k_pool, ptot, hcat, idxb, pooled, ptot);
  LAUNCH(k_conv1, (long long)BGR * C1K * L1LEN, pooled, c1w, c1b, y1, (long long)BGR * C1K * L1LEN);
  LAUNCH(k_mpool, (long long)BGR * C1K * L2LEN, y1, y2, (long long)BGR * C1K * L2LEN);
  LAUNCH(k_conv2, (long long)BGR * C2K * L3LEN, y2, c2w, c2b, y3, (long long)BGR * C2K * L3LEN);
  LAUNCH(k_dense, BGR * ODIM, y3, ow, ob, (float*)d_out, BGR * ODIM);
}
